// TGating_24970939859569
// MI455X (gfx1250) — matrix operand order repaired and checked by reading, not compiled, not run on hardware
//
#include <hip/hip_runtime.h>
#include <hip/hip_bf16.h>
#include <math.h>

// ---------------- types for WMMA ----------------
typedef __attribute__((ext_vector_type(16))) __bf16 v16bf;
typedef __attribute__((ext_vector_type(8)))  float  v8f;

union Frag { float4 q[2]; v16bf v; };

// ---------------- helpers ----------------
__device__ __forceinline__ float gelu_f(float v) {
    return 0.5f * v * (1.0f + erff(v * 0.70710678118654752f));
}
__device__ __forceinline__ float softplus_f(float x) {
    return x > 30.0f ? x : log1pf(expf(x));
}
__device__ __forceinline__ unsigned short f2bf(float f) {
    unsigned u = __float_as_uint(f);
    u += 0x7FFFu + ((u >> 16) & 1u);          // round-to-nearest-even
    return (unsigned short)(u >> 16);
}
__device__ __forceinline__ void sort24(float* a) {
    for (int i = 1; i < 24; i++) {
        float v = a[i]; int j = i - 1;
        while (j >= 0 && a[j] > v) { a[j + 1] = a[j]; j--; }
        a[j + 1] = v;
    }
}

#define TWO_PI 6.2831853071795864769f

// ============ K0: per-(b,lag,p) mean |xr - xr_shift| over H*W*C ============
__global__ void k_gating_diff(const float* __restrict__ x, float* __restrict__ dmat) {
    int blk = blockIdx.x;                // ((b*4 + l)*24 + p)
    int p = blk % 24;
    int l = (blk / 24) & 3;
    int b = blk / 96;
    const int LAG[4] = {1, 2, 4, 6};
    int ell = LAG[l];
    int ps = (p >= ell) ? (p - ell) : p;
    const float* xb = x + (size_t)b * 24 * 16384;
    float acc = 0.f;
    for (int i = threadIdx.x; i < 16384; i += 256) {
        float bp = (p <= 1) ? xb[i]
                 : (xb[(p - 2) * 16384 + i] + xb[(p - 1) * 16384 + i] + xb[p * 16384 + i]) * (1.f / 3.f);
        float xrp = xb[p * 16384 + i] - bp;
        float bq = (ps <= 1) ? xb[i]
                 : (xb[(ps - 2) * 16384 + i] + xb[(ps - 1) * 16384 + i] + xb[ps * 16384 + i]) * (1.f / 3.f);
        float xrq = xb[ps * 16384 + i] - bq;
        acc += fabsf(xrp - xrq);
    }
    __shared__ float red[256];
    red[threadIdx.x] = acc;
    __syncthreads();
    for (int s = 128; s > 0; s >>= 1) {
        if (threadIdx.x < s) red[threadIdx.x] += red[threadIdx.x + s];
        __syncthreads();
    }
    if (threadIdx.x == 0) dmat[blk] = red[0] * (1.f / 16384.f);
}

// ============ K1: medians/MAD/softplus/EMA/sigmoid -> att (B,24) ============
__global__ void k_att(const float* __restrict__ dmat, const float* __restrict__ pool,
                      float* __restrict__ att) {
    int b = threadIdx.x;
    if (b >= 8) return;
    float g[24];
    for (int p = 0; p < 24; p++) g[p] = 0.f;
    for (int l = 0; l < 4; l++) {
        float dv[24], tmp[24];
        for (int p = 0; p < 24; p++) { dv[p] = dmat[(b * 4 + l) * 24 + p]; tmp[p] = dv[p]; }
        sort24(tmp);
        float med = tmp[11];
        for (int p = 0; p < 24; p++) tmp[p] = fabsf(dv[p] - med);
        sort24(tmp);
        float mad = tmp[11];
        float denom = mad * 1.4826f + 1e-6f;
        for (int p = 0; p < 24; p++) g[p] += 0.25f * softplus_f((dv[p] - med) / denom);
    }
    float ge[24];
    float c = g[0]; ge[0] = c;
    for (int t = 1; t < 24; t++) { c = 0.6f * c + 0.4f * g[t]; ge[t] = c; }
    float mean = 0.f;
    for (int p = 0; p < 24; p++) mean += ge[p];
    mean *= (1.f / 24.f);
    float gat[24];
    for (int p = 0; p < 24; p++) gat[p] = 1.f / (1.f + expf(-1.5f * (ge[p] - mean)));
    // base_att = softmax(pool_param)
    float mx = -1e30f;
    for (int p = 0; p < 24; p++) mx = fmaxf(mx, pool[p]);
    float se = 0.f, ba[24];
    for (int p = 0; p < 24; p++) { ba[p] = expf(pool[p] - mx); se += ba[p]; }
    float w[24], mx2 = -1e30f;
    for (int p = 0; p < 24; p++) { w[p] = (ba[p] / se) * (1.f + gat[p]); mx2 = fmaxf(mx2, w[p]); }
    float s2 = 0.f;
    for (int p = 0; p < 24; p++) { w[p] = expf(w[p] - mx2); s2 += w[p]; }
    for (int p = 0; p < 24; p++) att[b * 24 + p] = w[p] / s2;
}

// ============ K2a: spatial mean -> noise -> time_emb[:,:,128:256] ============
__global__ void k_noise(const float* __restrict__ x, const float* __restrict__ noise_w,
                        const float* __restrict__ noise_b, float* __restrict__ time_emb) {
    int bp = blockIdx.x;                 // b*24 + p
    int c = threadIdx.x & 15, gidx = threadIdx.x >> 4;
    const float* xp = x + (size_t)bp * 16384;
    float s = 0.f;
    for (int n = gidx; n < 1024; n += 16) s += xp[n * 16 + c];
    __shared__ float red[256];
    red[threadIdx.x] = s;
    __syncthreads();
    for (int st = 8; st > 0; st >>= 1) {
        if (gidx < st) red[threadIdx.x] += red[threadIdx.x + st * 16];
        __syncthreads();
    }
    __shared__ float mean[16];
    if (threadIdx.x < 16) mean[threadIdx.x] = red[threadIdx.x] * (1.f / 1024.f);
    __syncthreads();
    if (threadIdx.x < 128) {
        int j = threadIdx.x;
        float a = noise_b[j];
        #pragma unroll
        for (int cc = 0; cc < 16; cc++) a += mean[cc] * noise_w[cc * 128 + j];
        time_emb[(size_t)bp * 256 + 128 + j] = a;
    }
}

// ============ K2b: periodic embedding -> time_emb[:,:,0:128] ============
__global__ void k_periodic(const float* __restrict__ pe_w1, const float* __restrict__ pe_b1,
                           const float* __restrict__ pe_w2, const float* __restrict__ pe_b2,
                           float* __restrict__ time_emb) {
    int p = blockIdx.x;                  // P = 24
    int j = threadIdx.x;                 // 128
    int k = j >> 6, q = j & 63;
    float period = (k == 0) ? 24.f : 72.f;
    float ph = TWO_PI * (float)p / period;
    float sn = sinf(ph), cs = cosf(ph);
    __shared__ float hid[128];
    hid[j] = gelu_f(sn * pe_w1[k * 128 + q] + cs * pe_w1[k * 128 + 64 + q] + pe_b1[k * 64 + q]);
    __syncthreads();
    float a = pe_b2[k * 64 + q];
    for (int t = 0; t < 64; t++) a += hid[k * 64 + t] * pe_w2[k * 4096 + t * 64 + q];
    for (int b = 0; b < 8; b++) time_emb[(size_t)((b * 24 + p)) * 256 + j] = a;
}

// ============ K3: DFT of (time_emb + bias), att-weighted te, time_vec ============
__global__ void k_tef(const float* __restrict__ time_emb, const float* __restrict__ in_proj_b,
                      const float* __restrict__ att, float* __restrict__ te_f,
                      float* __restrict__ zte, float* __restrict__ tvec) {
    int b = blockIdx.x;
    int c = threadIdx.x;
    float bc = in_proj_b[c];
    float acc[26];
    #pragma unroll
    for (int r = 0; r < 26; r++) acc[r] = 0.f;
    float m = 0.f, za = 0.f;
    for (int p = 0; p < 24; p++) {
        float te = time_emb[(size_t)(b * 24 + p) * 256 + c];
        m += te;
        za += att[b * 24 + p] * te;
        float tb = te + bc;
        #pragma unroll
        for (int r = 0; r < 13; r++) {
            float ang = TWO_PI * (float)(r * p) * (1.f / 24.f);
            acc[r]      +=  cosf(ang) * tb;
            acc[13 + r] += -sinf(ang) * tb;
        }
    }
    #pragma unroll
    for (int r = 0; r < 26; r++) te_f[(size_t)(b * 26 + r) * 256 + c] = acc[r];
    zte[b * 256 + c] = za + bc;
    tvec[b * 256 + c] = m * (1.f / 24.f);
}

// ============ K4: XF[b,n,r,cc] = sum_p Fb[r,p]*x[b,p,n,cc]  (r<26: DFT, r==26: att) ============
__global__ void k_xf(const float* __restrict__ x, const float* __restrict__ att,
                     float* __restrict__ XF) {
    __shared__ float Fb[27 * 24];
    int blk = blockIdx.x;                // 512 = 8b * 64
    int b = blk >> 6;
    int n0 = (blk & 63) << 4;
    for (int i = threadIdx.x; i < 27 * 24; i += 256) {
        int r = i / 24, p = i % 24;
        float v;
        if (r < 13)      v =  cosf(TWO_PI * (float)(r * p) * (1.f / 24.f));
        else if (r < 26) v = -sinf(TWO_PI * (float)((r - 13) * p) * (1.f / 24.f));
        else             v = att[b * 24 + p];
        Fb[i] = v;
    }
    __syncthreads();
    int n = n0 + (threadIdx.x >> 4);
    int cc = threadIdx.x & 15;
    float acc[27];
    #pragma unroll
    for (int r = 0; r < 27; r++) acc[r] = 0.f;
    const float* xb = x + (((size_t)b * 24) * 1024 + n) * 16 + cc;
    for (int p = 0; p < 24; p++) {
        float v = xb[(size_t)p * 16384];
        #pragma unroll
        for (int r = 0; r < 27; r++) acc[r] += Fb[r * 24 + p] * v;
    }
    float* o = XF + (((size_t)b * 1024 + n) * 27) * 16 + cc;
    #pragma unroll
    for (int r = 0; r < 27; r++) o[r * 16] = acc[r];
}

// ============ K5a: Wkr/Wki[k,cc,o] = Win^T folded into sfe_w1 ============
__global__ void k_wk(const float* __restrict__ in_proj_w, const float* __restrict__ sfe_w1,
                     float* __restrict__ Wkr, float* __restrict__ Wki) {
    __shared__ float wlds[4096];         // in_proj_w (16x256)
    for (int i = threadIdx.x; i < 4096; i += 256) wlds[i] = in_proj_w[i];
    __syncthreads();
    int blk = blockIdx.x;                // 26: k*2 + ri
    int k = blk >> 1, ri = blk & 1;
    int o = threadIdx.x;
    int eoff = ri ? 256 : 0;
    float acc[16];
    #pragma unroll
    for (int cc = 0; cc < 16; cc++) acc[cc] = 0.f;
    for (int c = 0; c < 256; c++) {
        float s = sfe_w1[((size_t)(k * 512 + eoff + c)) * 256 + o];
        #pragma unroll
        for (int cc = 0; cc < 16; cc++) acc[cc] += wlds[cc * 256 + c] * s;
    }
    float* dst = (ri ? Wki : Wkr) + (size_t)(k * 16) * 256 + o;
    #pragma unroll
    for (int cc = 0; cc < 16; cc++) dst[(size_t)cc * 256] = acc[cc];
}

// ============ K5b: bias1[b,k,o] = sfe_b1 + te_f part ============
__global__ void k_bias1(const float* __restrict__ sfe_w1, const float* __restrict__ sfe_b1,
                        const float* __restrict__ te_f, float* __restrict__ bias1) {
    int blk = blockIdx.x;                // b*13 + k
    int b = blk / 13, k = blk % 13;
    int o = threadIdx.x;
    float acc = sfe_b1[k * 256 + o];
    const float* tr = te_f + (size_t)(b * 26 + k) * 256;
    const float* ti = te_f + (size_t)(b * 26 + 13 + k) * 256;
    const float* w = sfe_w1 + (size_t)k * 512 * 256;
    for (int c = 0; c < 256; c++)
        acc += tr[c] * w[(size_t)c * 256 + o] + ti[c] * w[(size_t)(256 + c) * 256 + o];
    bias1[(size_t)blk * 256 + o] = acc;
}

// ============ K5c: BW[o][kk] = (sfe_w2[k] @ comb_w1[k-block])^T in WMMA-B layout ============
__global__ void k_bw(const float* __restrict__ sfe_w2, const float* __restrict__ comb_w1,
                     unsigned short* __restrict__ BW) {
    int kk = blockIdx.x;                 // 0..3327 = k*256 + c
    int k = kk >> 8, c = kk & 255;
    int o = threadIdx.x;
    const float* w2 = sfe_w2 + ((size_t)k * 256 + c) * 256;
    float acc = 0.f;
    for (int c2 = 0; c2 < 256; c2++)
        acc += w2[c2] * comb_w1[((size_t)(k * 256 + c2)) * 256 + o];
    BW[(size_t)o * 3328 + kk] = f2bf(acc);
}

// ============ K5d: biasH1[o] = comb_b1 + sum sfe_b2 @ comb_w1 ============
__global__ void k_biash1(const float* __restrict__ sfe_b2, const float* __restrict__ comb_w1,
                         const float* __restrict__ comb_b1, float* __restrict__ biasH1) {
    int o = threadIdx.x;
    float acc = comb_b1[o];
    for (int kk = 0; kk < 3328; kk++) acc += sfe_b2[kk] * comb_w1[(size_t)kk * 256 + o];
    biasH1[o] = acc;
}

// ============ K5e: W2T[n][k] = comb_w2[k][n] (bf16, WMMA-B layout) ============
__global__ void k_w2t(const float* __restrict__ comb_w2, unsigned short* __restrict__ W2T) {
    int n = blockIdx.x, k = threadIdx.x;
    W2T[(size_t)n * 256 + k] = f2bf(comb_w2[(size_t)k * 256 + n]);
}

// ============ K6: P1[b,n,k,o] = gelu(rank-32 contraction + bias1), bf16 ============
__global__ void k_p1(const float* __restrict__ XF, const float* __restrict__ Wkr,
                     const float* __restrict__ Wki, const float* __restrict__ bias1,
                     unsigned short* __restrict__ P1) {
    int bn = blockIdx.x;                 // 0..8191
    int b = bn >> 10;
    __shared__ float xf[432];
    for (int i = threadIdx.x; i < 432; i += 256) xf[i] = XF[(size_t)bn * 432 + i];
    __syncthreads();
    int o = threadIdx.x;
    unsigned short* dst = P1 + (size_t)bn * 3328 + o;
    const float* b1 = bias1 + (size_t)(b * 13) * 256 + o;
    for (int k = 0; k < 13; k++) {
        float acc = b1[(size_t)k * 256];
        const float* wr = Wkr + (size_t)(k * 16) * 256 + o;
        const float* wi = Wki + (size_t)(k * 16) * 256 + o;
        #pragma unroll
        for (int cc = 0; cc < 16; cc++)
            acc += xf[k * 16 + cc] * wr[(size_t)cc * 256] + xf[(13 + k) * 16 + cc] * wi[(size_t)cc * 256];
        dst[(size_t)k * 256] = f2bf(gelu_f(acc));
    }
}

// ============ K7: WMMA GEMM1: G = gelu(P1 @ M + biasH1), M=8192 K=3328 N=256 ============
__global__ void k_gemm1(const unsigned short* __restrict__ P1, const unsigned short* __restrict__ BW,
                        const float* __restrict__ biasH1, unsigned short* __restrict__ G) {
    int wave = threadIdx.x >> 5;
    int lane = threadIdx.x & 31;
    int rowBase = blockIdx.x * 128 + wave * 16;
    int colBase = blockIdx.y * 64;
    int lrow = lane & 15;
    int hi = lane >> 4;                  // 0: K 0-7,16-23; 1: K 8-15,24-31 of each 32-K step
    v8f zero = {0.f, 0.f, 0.f, 0.f, 0.f, 0.f, 0.f, 0.f};
    v8f acc[4];
    #pragma unroll
    for (int j = 0; j < 4; j++) acc[j] = zero;
    const unsigned short* Aptr = P1 + (size_t)(rowBase + lrow) * 3328 + hi * 8;
    for (int kk = 0; kk < 3328; kk += 32) {
        __builtin_prefetch((const void*)(Aptr + kk + 512), 0, 1);
        Frag a;
        a.q[0] = *(const float4*)(Aptr + kk);        // K = kk+ksel .. +7
        a.q[1] = *(const float4*)(Aptr + kk + 16);   // K = kk+ksel+16 .. +23
        #pragma unroll
        for (int j = 0; j < 4; j++) {
            const unsigned short* bp = BW + (size_t)(colBase + 16 * j + lrow) * 3328 + hi * 8 + kk;
            Frag bq;
            bq.q[0] = *(const float4*)(bp);
            bq.q[1] = *(const float4*)(bp + 16);
            acc[j] = __builtin_amdgcn_wmma_f32_16x16x32_bf16(
                false, a.v, false, bq.v, (short)0, acc[j], false, false);
        }
    }
    #pragma unroll
    for (int j = 0; j < 4; j++) {
        int n = colBase + 16 * j + lrow;
        float bh = biasH1[n];
        #pragma unroll
        for (int v = 0; v < 8; v++) {
            int m = rowBase + v + hi * 8;
            G[(size_t)m * 256 + n] = f2bf(gelu_f(acc[j][v] + bh));
        }
    }
}

// ============ K8: WMMA GEMM2 + z epilogue: z = Xatt@Win + zte + 0.3*(G@W2 + b2) ============
__global__ void k_gemm2(const unsigned short* __restrict__ G, const unsigned short* __restrict__ W2T,
                        const float* __restrict__ comb_b2, const float* __restrict__ XF,
                        const float* __restrict__ zte, const float* __restrict__ in_proj_w,
                        float* __restrict__ zout) {
    int wave = threadIdx.x >> 5;
    int lane = threadIdx.x & 31;
    int rowBase = blockIdx.x * 128 + wave * 16;
    int colBase = blockIdx.y * 64;
    int lrow = lane & 15;
    int hi = lane >> 4;
    v8f zero = {0.f, 0.f, 0.f, 0.f, 0.f, 0.f, 0.f, 0.f};
    v8f acc[4];
    #pragma unroll
    for (int j = 0; j < 4; j++) acc[j] = zero;
    const unsigned short* Aptr = G + (size_t)(rowBase + lrow) * 256 + hi * 8;
    #pragma unroll
    for (int kk = 0; kk < 256; kk += 32) {
        Frag a;
        a.q[0] = *(const float4*)(Aptr + kk);
        a.q[1] = *(const float4*)(Aptr + kk + 16);
        #pragma unroll
        for (int j = 0; j < 4; j++) {
            const unsigned short* bp = W2T + (size_t)(colBase + 16 * j + lrow) * 256 + hi * 8 + kk;
            Frag bq;
            bq.q[0] = *(const float4*)(bp);
            bq.q[1] = *(const float4*)(bp + 16);
            acc[j] = __builtin_amdgcn_wmma_f32_16x16x32_bf16(
                false, a.v, false, bq.v, (short)0, acc[j], false, false);
        }
    }
    #pragma unroll
    for (int j = 0; j < 4; j++) {
        int n = colBase + 16 * j + lrow;
        float cb = comb_b2[n];
        float win[16];
        #pragma unroll
        for (int cc = 0; cc < 16; cc++) win[cc] = in_proj_w[cc * 256 + n];
        #pragma unroll
        for (int v = 0; v < 8; v++) {
            int m = rowBase + v + hi * 8;
            int b = m >> 10;
            float hf = acc[j][v] + cb;
            const float* xa = XF + (size_t)m * 432 + 416;   // r==26 (att row)
            float zl = zte[b * 256 + n];
            #pragma unroll
            for (int cc = 0; cc < 16; cc++) zl += xa[cc] * win[cc];
            zout[(size_t)m * 256 + n] = zl + 0.3f * hf;
        }
    }
}

// ================= host launcher =================
extern "C" void kernel_launch(void* const* d_in, const int* in_sizes, int n_in,
                              void* d_out, int out_size, void* d_ws, size_t ws_size,
                              hipStream_t stream) {
    const float* x         = (const float*)d_in[0];
    const float* in_proj_w = (const float*)d_in[1];
    const float* in_proj_b = (const float*)d_in[2];
    const float* pe_w1     = (const float*)d_in[3];
    const float* pe_b1     = (const float*)d_in[4];
    const float* pe_w2     = (const float*)d_in[5];
    const float* pe_b2     = (const float*)d_in[6];
    const float* noise_w   = (const float*)d_in[7];
    const float* noise_b   = (const float*)d_in[8];
    const float* sfe_w1    = (const float*)d_in[9];
    const float* sfe_b1    = (const float*)d_in[10];
    const float* sfe_w2    = (const float*)d_in[11];
    const float* sfe_b2    = (const float*)d_in[12];
    const float* comb_w1   = (const float*)d_in[13];
    const float* comb_b1   = (const float*)d_in[14];
    const float* comb_w2   = (const float*)d_in[15];
    const float* comb_b2   = (const float*)d_in[16];
    const float* pool      = (const float*)d_in[17];
    (void)in_sizes; (void)n_in; (void)out_size; (void)ws_size;

    float* zout = (float*)d_out;
    float* tvec = zout + (size_t)8 * 1024 * 256;

    char* w = (char*)d_ws;
    size_t off = 0;
    auto A = [&](size_t bytes) -> char* {
        char* p = w + off;
        off = (off + bytes + 255) & ~(size_t)255;
        return p;
    };
    float* dmat      = (float*)A(768 * 4);
    float* att       = (float*)A(192 * 4);
    float* time_emb  = (float*)A(49152 * 4);
    float* te_f      = (float*)A(53248 * 4);
    float* zte       = (float*)A(2048 * 4);
    float* XF        = (float*)A(3538944ull * 4);
    float* Wkr       = (float*)A(53248 * 4);
    float* Wki       = (float*)A(53248 * 4);
    float* bias1     = (float*)A(26624 * 4);
    float* biasH1    = (float*)A(256 * 4);
    unsigned short* P1  = (unsigned short*)A(27262976ull * 2);
    unsigned short* BW  = (unsigned short*)A(851968ull * 2);
    unsigned short* W2T = (unsigned short*)A(65536ull * 2);
    unsigned short* G   = (unsigned short*)A(2097152ull * 2);

    k_gating_diff<<<768, 256, 0, stream>>>(x, dmat);
    k_att<<<1, 32, 0, stream>>>(dmat, pool, att);
    k_noise<<<192, 256, 0, stream>>>(x, noise_w, noise_b, time_emb);
    k_periodic<<<24, 128, 0, stream>>>(pe_w1, pe_b1, pe_w2, pe_b2, time_emb);
    k_tef<<<8, 256, 0, stream>>>(time_emb, in_proj_b, att, te_f, zte, tvec);
    k_xf<<<512, 256, 0, stream>>>(x, att, XF);
    k_wk<<<26, 256, 0, stream>>>(in_proj_w, sfe_w1, Wkr, Wki);
    k_bias1<<<104, 256, 0, stream>>>(sfe_w1, sfe_b1, te_f, bias1);
    k_bw<<<3328, 256, 0, stream>>>(sfe_w2, comb_w1, BW);
    k_biash1<<<1, 256, 0, stream>>>(sfe_b2, comb_w1, comb_b1, biasH1);
    k_w2t<<<256, 256, 0, stream>>>(comb_w2, W2T);
    k_p1<<<8192, 256, 0, stream>>>(XF, Wkr, Wki, bias1, P1);
    k_gemm1<<<dim3(64, 4), 256, 0, stream>>>(P1, BW, biasH1, G);
    k_gemm2<<<dim3(64, 4), 256, 0, stream>>>(G, W2T, comb_b2, XF, zte, in_proj_w, zout);
}
